// H2GCN_57097295233465
// MI455X (gfx1250) — compile-verified, matrix operand order checked by reading, not run on hardware
//
#include <hip/hip_runtime.h>
#include <hip/hip_bf16.h>

#define N_NODES 50000
#define IN_FEAT 500
#define HIDDEN  128
#define N_CLASS 16
#define E1_CNT  1000000
#define E2_CNT  1500000
#define M_TILES 3125          // 50000 / 16
#define EMB_CH  16            // K chunks for embed (512 padded / 32)
#define CLS_CH  28            // K chunks for classify (896 / 32)
#define NB_SCAN 196           // ceil(50000/256)

typedef __attribute__((ext_vector_type(16))) __bf16 v16bf;
typedef __attribute__((ext_vector_type(8)))  float  v8f;

__device__ __forceinline__ float4 ld4(const float* p) { return *(const float4*)p; }

// pack 16 fp32 -> v16bf (lowers to v_cvt_pk_bf16_f32 pairs)
__device__ __forceinline__ v16bf pack16(float4 x0, float4 x1, float4 x2, float4 x3) {
    v16bf r;
    r[0]=(__bf16)x0.x;  r[1]=(__bf16)x0.y;  r[2]=(__bf16)x0.z;  r[3]=(__bf16)x0.w;
    r[4]=(__bf16)x1.x;  r[5]=(__bf16)x1.y;  r[6]=(__bf16)x1.z;  r[7]=(__bf16)x1.w;
    r[8]=(__bf16)x2.x;  r[9]=(__bf16)x2.y;  r[10]=(__bf16)x2.z; r[11]=(__bf16)x2.w;
    r[12]=(__bf16)x3.x; r[13]=(__bf16)x3.y; r[14]=(__bf16)x3.z; r[15]=(__bf16)x3.w;
    return r;
}

// ---------------------------------------------------------------------------
// Weight pre-swizzle into exact WMMA B-fragment lane order (bf16, K-padded).
// element ((n*NCH + c)*2 + half)*16 + j  <->  k = c*32 + (j>>3)*16 + half*8 + (j&7)
// ---------------------------------------------------------------------------
__global__ void swizzle_w_embed_kernel(const float* __restrict__ W, __bf16* __restrict__ Wswz) {
    int idx = blockIdx.x * blockDim.x + threadIdx.x;        // 128*16*2*16 = 65536
    int j = idx & 15, half = (idx >> 4) & 1, c = (idx >> 5) & 15, n = idx >> 9;
    int k = c * 32 + ((j >> 3) << 4) + (half << 3) + (j & 7);
    float v = (k < IN_FEAT) ? W[k * HIDDEN + n] : 0.f;
    Wswz[idx] = (__bf16)v;
}

__global__ void swizzle_w_cls_kernel(const float* __restrict__ Wc, __bf16* __restrict__ Wcswz) {
    int idx = blockIdx.x * blockDim.x + threadIdx.x;        // 16*28*2*16 = 14336
    int j = idx & 15, half = (idx >> 4) & 1, c = (idx >> 5) % CLS_CH, n = idx / (CLS_CH * 32);
    int k = c * 32 + ((j >> 3) << 4) + (half << 3) + (j & 7);
    Wcswz[idx] = (__bf16)Wc[k * N_CLASS + n];
}

// ---------------------------------------------------------------------------
// small utility kernels
// ---------------------------------------------------------------------------
__global__ void zero4_kernel(float4* __restrict__ p, int n4) {
    int i = blockIdx.x * blockDim.x + threadIdx.x;
    if (i < n4) p[i] = make_float4(0.f, 0.f, 0.f, 0.f);
}

__global__ void relu4_kernel(float4* __restrict__ p, int n4) {
    int i = blockIdx.x * blockDim.x + threadIdx.x;
    if (i < n4) {
        float4 v = p[i];
        v.x = fmaxf(v.x, 0.f); v.y = fmaxf(v.y, 0.f);
        v.z = fmaxf(v.z, 0.f); v.w = fmaxf(v.w, 0.f);
        p[i] = v;
    }
}

__global__ void copy_int_kernel(const int* __restrict__ src, int* __restrict__ dst, int n) {
    int i = blockIdx.x * blockDim.x + threadIdx.x;
    if (i < n) dst[i] = src[i];
}

// ---------------------------------------------------------------------------
// CSR build: histogram -> 3-phase exclusive scan -> scatter fill
// ---------------------------------------------------------------------------
__global__ void hist_kernel(const int* __restrict__ rows, int n, int* __restrict__ cnt) {
    int i = blockIdx.x * blockDim.x + threadIdx.x;
    if (i < n) atomicAdd(&cnt[rows[i]], 1);
}

__global__ void block_sum_kernel(const int* __restrict__ cnt, int* __restrict__ bsum, int n) {
    __shared__ int s[256];
    int t = threadIdx.x, i = blockIdx.x * 256 + t;
    s[t] = (i < n) ? cnt[i] : 0;
    __syncthreads();
    for (int off = 128; off > 0; off >>= 1) {
        if (t < off) s[t] += s[t + off];
        __syncthreads();
    }
    if (t == 0) bsum[blockIdx.x] = s[0];
}

__global__ void scan_small_kernel(int* __restrict__ bsum, int nb) {
    __shared__ int s[256];
    int t = threadIdx.x;
    int v = (t < nb) ? bsum[t] : 0;
    s[t] = v;
    __syncthreads();
    for (int off = 1; off < 256; off <<= 1) {
        int x = (t >= off) ? s[t - off] : 0;
        __syncthreads();
        s[t] += x;
        __syncthreads();
    }
    if (t < nb) bsum[t] = s[t] - v;    // exclusive block offsets
}

__global__ void scan_write_kernel(const int* __restrict__ cnt, const int* __restrict__ bsum,
                                  int* __restrict__ rp, int n) {
    __shared__ int s[256];
    int t = threadIdx.x, i = blockIdx.x * 256 + t;
    int v = (i < n) ? cnt[i] : 0;
    s[t] = v;
    __syncthreads();
    for (int off = 1; off < 256; off <<= 1) {
        int x = (t >= off) ? s[t - off] : 0;
        __syncthreads();
        s[t] += x;
        __syncthreads();
    }
    int base = bsum[blockIdx.x];
    if (i < n) rp[i] = base + s[t] - v;
    if (i == n - 1) rp[n] = base + s[t];
}

__global__ void fill_csr_kernel(const int* __restrict__ rows, const int* __restrict__ cols,
                                const float* __restrict__ vals, int n,
                                int* __restrict__ cur, uint2* __restrict__ ecv) {
    int i = blockIdx.x * blockDim.x + threadIdx.x;
    if (i < n) {
        int pos = atomicAdd(&cur[rows[i]], 1);
        ecv[pos] = make_uint2((unsigned)cols[i], __float_as_uint(vals[i]));
    }
}

// ---------------------------------------------------------------------------
// Gather-only CSR SpMM with fused ReLU: one wave per output row, register
// accumulation, single store per element (no atomics, no zero/relu passes).
// Edge (col,val) pairs loaded 32-wide, broadcast via shfl; gathers L2-resident.
// ---------------------------------------------------------------------------
template <int D>
__global__ void __launch_bounds__(256)
spmm_csr_kernel(const int* __restrict__ rp, const uint2* __restrict__ ecv,
                const float* __restrict__ R, float* __restrict__ out,
                int out_stride, int out_off) {
    int row  = (blockIdx.x * blockDim.x + threadIdx.x) >> 5;
    int lane = threadIdx.x & 31;
    if (row >= N_NODES) return;
    row = __builtin_amdgcn_readfirstlane(row);     // wave-uniform -> scalar loads
    int beg = rp[row], end = rp[row + 1];

    constexpr int NA = D / 32;                     // 4 (d=128) or 8 (d=256) per lane
    float acc[NA];
#pragma unroll
    for (int q = 0; q < NA; ++q) acc[q] = 0.f;

    for (int i = beg; i < end; i += 32) {
        int e = i + lane;
        uint2 cv = make_uint2(0u, 0u);
        if (e < end) cv = ecv[e];
        int cnt = end - i; if (cnt > 32) cnt = 32;
        for (int t = 0; t < cnt; ++t) {
            int   col = __shfl((int)cv.x, t, 32);
            float val = __uint_as_float(__shfl((int)cv.y, t, 32));
            const float* src = R + (size_t)col * D + lane * 4;
#pragma unroll
            for (int q = 0; q < NA; q += 4) {
                float4 xv = ld4(src + (q >> 2) * 128);
                acc[q + 0] += val * xv.x;
                acc[q + 1] += val * xv.y;
                acc[q + 2] += val * xv.z;
                acc[q + 3] += val * xv.w;
            }
        }
    }

    float* dst = out + (size_t)row * out_stride + out_off + lane * 4;
#pragma unroll
    for (int q = 0; q < NA; q += 4) {
        float4 o;
        o.x = fmaxf(acc[q + 0], 0.f); o.y = fmaxf(acc[q + 1], 0.f);
        o.z = fmaxf(acc[q + 2], 0.f); o.w = fmaxf(acc[q + 3], 0.f);
        *(float4*)(dst + (q >> 2) * 128) = o;
    }
}

// ---------------------------------------------------------------------------
// Fallback atomic scatter SpMM (used only if workspace too small for CSR)
// ---------------------------------------------------------------------------
__global__ void __launch_bounds__(256)
spmm_scatter_kernel(const int* __restrict__ rows, const int* __restrict__ cols,
                    const float* __restrict__ vals, int nedges,
                    const float* __restrict__ R, int d,
                    float* __restrict__ out, int out_stride, int out_off) {
    int wid  = (blockIdx.x * blockDim.x + threadIdx.x) >> 5;
    int lane = threadIdx.x & 31;
    if (wid >= nedges) return;
    wid = __builtin_amdgcn_readfirstlane(wid);
    int   rrow = rows[wid];
    int   rcol = cols[wid];
    float v    = vals[wid];
    const float* src = R + (size_t)rcol * d;
    float* dst = out + (size_t)rrow * out_stride + out_off;
    for (int j = lane * 4; j < d; j += 128) {
        float4 x = ld4(src + j);
        atomicAdd(dst + j + 0, v * x.x);
        atomicAdd(dst + j + 1, v * x.y);
        atomicAdd(dst + j + 2, v * x.z);
        atomicAdd(dst + j + 3, v * x.w);
    }
}

// ---------------------------------------------------------------------------
// Embed GEMM: R0 = relu(X[50000,500] @ W[500,128]) via bf16 WMMA.
// One wave per 16-row tile; 8 accumulators cover N=128 so X is read once.
// ---------------------------------------------------------------------------
__global__ void __launch_bounds__(256)
embed_wmma_kernel(const float* __restrict__ X, const __bf16* __restrict__ Wswz,
                  float* __restrict__ R0) {
    int tile = blockIdx.x * 8 + (threadIdx.x >> 5);
    if (tile >= M_TILES) return;
    int lane = threadIdx.x & 31;
    int mr   = lane & 15;
    int half = lane >> 4;
    const float* xrow = X + (size_t)(tile * 16 + mr) * IN_FEAT;

    v8f acc[8];
#pragma unroll
    for (int t = 0; t < 8; ++t) acc[t] = (v8f){0.f,0.f,0.f,0.f,0.f,0.f,0.f,0.f};

    for (int c = 0; c < EMB_CH - 1; ++c) {       // k0 = 0..479, guard-free
        const float* pa = xrow + c * 32 + (half << 3);
        v16bf a = pack16(ld4(pa), ld4(pa + 4), ld4(pa + 16), ld4(pa + 20));
#pragma unroll
        for (int t = 0; t < 8; ++t) {
            int n = t * 16 + mr;
            v16bf b = *(const v16bf*)(Wswz + (size_t)((n * EMB_CH + c) * 2 + half) * 16);
            acc[t] = __builtin_amdgcn_wmma_f32_16x16x32_bf16(
                false, a, false, b, (short)0, acc[t], false, false);
        }
    }
    {   // tail chunk c = 15: k 480..495 full, 496..499 valid only for half==0
        const float* pa = xrow + 480 + (half << 3);
        float4 x0 = ld4(pa), x1 = ld4(pa + 4);
        float4 x2 = make_float4(0.f, 0.f, 0.f, 0.f);
        if (half == 0) x2 = ld4(xrow + 496);
        v16bf a = pack16(x0, x1, x2, make_float4(0.f, 0.f, 0.f, 0.f));
#pragma unroll
        for (int t = 0; t < 8; ++t) {
            int n = t * 16 + mr;
            v16bf b = *(const v16bf*)(Wswz + (size_t)((n * EMB_CH + 15) * 2 + half) * 16);
            acc[t] = __builtin_amdgcn_wmma_f32_16x16x32_bf16(
                false, a, false, b, (short)0, acc[t], false, false);
        }
    }

    int mo = tile * 16 + (half << 3);
#pragma unroll
    for (int t = 0; t < 8; ++t) {
        int n = t * 16 + mr;
#pragma unroll
        for (int r = 0; r < 8; ++r) {
            float v = acc[t][r];
            R0[(size_t)(mo + r) * HIDDEN + n] = v > 0.f ? v : 0.f;
        }
    }
}

// ---------------------------------------------------------------------------
// Classify + softmax: out = softmax([rs0|rs1|rs2][50000,896] @ Wc[896,16]).
// ---------------------------------------------------------------------------
__global__ void __launch_bounds__(256)
classify_wmma_kernel(const float* __restrict__ rs0, const float* __restrict__ rs1,
                     const float* __restrict__ rs2, const __bf16* __restrict__ Wcswz,
                     float* __restrict__ out) {
    int tile = blockIdx.x * 8 + (threadIdx.x >> 5);
    if (tile >= M_TILES) return;
    int lane = threadIdx.x & 31;
    int mr   = lane & 15;
    int half = lane >> 4;
    int m    = tile * 16 + mr;

    v8f acc = (v8f){0.f,0.f,0.f,0.f,0.f,0.f,0.f,0.f};

#pragma unroll
    for (int c = 0; c < CLS_CH; ++c) {
        int k0 = c * 32;
        const float* base; int stride, koff;
        if (k0 < 128)      { base = rs0; stride = 128; koff = k0;       }
        else if (k0 < 384) { base = rs1; stride = 256; koff = k0 - 128; }
        else               { base = rs2; stride = 512; koff = k0 - 384; }
        const float* pa = base + (size_t)m * stride + koff + (half << 3);
        v16bf a = pack16(ld4(pa), ld4(pa + 4), ld4(pa + 16), ld4(pa + 20));
        v16bf b = *(const v16bf*)(Wcswz + (size_t)((mr * CLS_CH + c) * 2 + half) * 16);
        acc = __builtin_amdgcn_wmma_f32_16x16x32_bf16(
            false, a, false, b, (short)0, acc, false, false);
    }

#pragma unroll
    for (int r = 0; r < 8; ++r) {
        float v  = acc[r];
        float mx = v;
#pragma unroll
        for (int s = 1; s < 16; s <<= 1)
            mx = fmaxf(mx, __shfl_xor(mx, s, 32));
        float e   = __expf(v - mx);
        float sum = e;
#pragma unroll
        for (int s = 1; s < 16; s <<= 1)
            sum += __shfl_xor(sum, s, 32);
        int mo = tile * 16 + (half << 3) + r;
        out[(size_t)mo * N_CLASS + mr] = e / sum;
    }
}

// ---------------------------------------------------------------------------
// Host-side orchestration (graph-capture safe: all launches on `stream`).
// ---------------------------------------------------------------------------
extern "C" void kernel_launch(void* const* d_in, const int* in_sizes, int n_in,
                              void* d_out, int out_size, void* d_ws, size_t ws_size,
                              hipStream_t stream) {
    (void)in_sizes; (void)n_in; (void)out_size;
    const float* x   = (const float*)d_in[0];
    const int*   a1r = (const int*)  d_in[1];
    const int*   a1c = (const int*)  d_in[2];
    const float* a1v = (const float*)d_in[3];
    const int*   a2r = (const int*)  d_in[4];
    const int*   a2c = (const int*)  d_in[5];
    const float* a2v = (const float*)d_in[6];
    const float* we  = (const float*)d_in[7];
    const float* wc  = (const float*)d_in[8];

    char* base = (char*)d_ws;
    // byte offsets (all 16B aligned)
    const size_t OFF_RS0   = 0;                        // 50000*128 f32 = 25.6e6 B
    const size_t OFF_RS1   = 25600000;                 // 50000*256 f32 = 51.2e6 B
    const size_t OFF_RS2   = 76800000;                 // 50000*512 f32 = 102.4e6 B
    const size_t OFF_WSWZ  = 179200000;                // 65536 bf16
    const size_t OFF_WCSWZ = OFF_WSWZ  + 131072;       // 14336 bf16
    const size_t OFF_RP1   = OFF_WCSWZ + 28672;        // 50001 i32 (padded)
    const size_t OFF_RP2   = OFF_RP1 + 200016;
    const size_t OFF_CUR1  = OFF_RP2 + 200016;         // 50000 i32
    const size_t OFF_CUR2  = OFF_CUR1 + 200000;
    const size_t OFF_BSUM  = OFF_CUR2 + 200000;        // 256 i32
    const size_t OFF_ECV1  = OFF_BSUM + 1024;          // 1.0M uint2
    const size_t OFF_ECV2  = OFF_ECV1 + 8000000;       // 1.5M uint2
    const size_t CSR_NEED  = OFF_ECV2 + 12000000;      // ~200.2 MB total

    float*  rs0   = (float*)(base + OFF_RS0);
    float*  rs1   = (float*)(base + OFF_RS1);
    float*  rs2   = (float*)(base + OFF_RS2);
    __bf16* wswz  = (__bf16*)(base + OFF_WSWZ);
    __bf16* wcswz = (__bf16*)(base + OFF_WCSWZ);
    float*  outp  = (float*)d_out;

    // weight pre-swizzle (tiny, L2-resident afterwards)
    swizzle_w_embed_kernel<<<256, 256, 0, stream>>>(we, wswz);
    swizzle_w_cls_kernel<<<56, 256, 0, stream>>>(wc, wcswz);

    // rs0 = relu(x @ w_embed)
    embed_wmma_kernel<<<(M_TILES + 7) / 8, 256, 0, stream>>>(x, wswz, rs0);

    if (ws_size >= CSR_NEED) {
        // ---- CSR path: no fp32 atomics, fused relu, no zero passes ----
        int*   rp1  = (int*)(base + OFF_RP1);
        int*   rp2  = (int*)(base + OFF_RP2);
        int*   cur1 = (int*)(base + OFF_CUR1);
        int*   cur2 = (int*)(base + OFF_CUR2);
        int*   bsum = (int*)(base + OFF_BSUM);
        uint2* ecv1 = (uint2*)(base + OFF_ECV1);
        uint2* ecv2 = (uint2*)(base + OFF_ECV2);

        // build CSR for A1 (cur1 doubles as count buffer)
        zero4_kernel<<<12500 / 256 + 1, 256, 0, stream>>>((float4*)cur1, 12500);
        hist_kernel<<<(E1_CNT + 255) / 256, 256, 0, stream>>>(a1r, E1_CNT, cur1);
        block_sum_kernel<<<NB_SCAN, 256, 0, stream>>>(cur1, bsum, N_NODES);
        scan_small_kernel<<<1, 256, 0, stream>>>(bsum, NB_SCAN);
        scan_write_kernel<<<NB_SCAN, 256, 0, stream>>>(cur1, bsum, rp1, N_NODES);
        copy_int_kernel<<<(N_NODES + 255) / 256, 256, 0, stream>>>(rp1, cur1, N_NODES);
        fill_csr_kernel<<<(E1_CNT + 255) / 256, 256, 0, stream>>>(a1r, a1c, a1v, E1_CNT, cur1, ecv1);
        // build CSR for A2
        zero4_kernel<<<12500 / 256 + 1, 256, 0, stream>>>((float4*)cur2, 12500);
        hist_kernel<<<(E2_CNT + 255) / 256, 256, 0, stream>>>(a2r, E2_CNT, cur2);
        block_sum_kernel<<<NB_SCAN, 256, 0, stream>>>(cur2, bsum, N_NODES);
        scan_small_kernel<<<1, 256, 0, stream>>>(bsum, NB_SCAN);
        scan_write_kernel<<<NB_SCAN, 256, 0, stream>>>(cur2, bsum, rp2, N_NODES);
        copy_int_kernel<<<(N_NODES + 255) / 256, 256, 0, stream>>>(rp2, cur2, N_NODES);
        fill_csr_kernel<<<(E2_CNT + 255) / 256, 256, 0, stream>>>(a2r, a2c, a2v, E2_CNT, cur2, ecv2);

        // iteration 1: rs1 = relu([A1@rs0 | A2@rs0])
        spmm_csr_kernel<128><<<6250, 256, 0, stream>>>(rp1, ecv1, rs0, rs1, 256, 0);
        spmm_csr_kernel<128><<<6250, 256, 0, stream>>>(rp2, ecv2, rs0, rs1, 256, 128);
        // iteration 2: rs2 = relu([A1@rs1 | A2@rs1])
        spmm_csr_kernel<256><<<6250, 256, 0, stream>>>(rp1, ecv1, rs1, rs2, 512, 0);
        spmm_csr_kernel<256><<<6250, 256, 0, stream>>>(rp2, ecv2, rs1, rs2, 512, 256);
    } else {
        // ---- fallback: atomic scatter ----
        zero4_kernel<<<37500, 256, 0, stream>>>((float4*)rs1, 9600000);
        spmm_scatter_kernel<<<E1_CNT / 8, 256, 0, stream>>>(a1r, a1c, a1v, E1_CNT,
                                                            rs0, HIDDEN, rs1, 256, 0);
        spmm_scatter_kernel<<<E2_CNT / 8, 256, 0, stream>>>(a2r, a2c, a2v, E2_CNT,
                                                            rs0, HIDDEN, rs1, 256, 128);
        relu4_kernel<<<12500, 256, 0, stream>>>((float4*)rs1, 3200000);
        spmm_scatter_kernel<<<E1_CNT / 8, 256, 0, stream>>>(a1r, a1c, a1v, E1_CNT,
                                                            rs1, 256, rs2, 512, 0);
        spmm_scatter_kernel<<<E2_CNT / 8, 256, 0, stream>>>(a2r, a2c, a2v, E2_CNT,
                                                            rs1, 256, rs2, 512, 256);
        relu4_kernel<<<25000, 256, 0, stream>>>((float4*)rs2, 6400000);
    }

    // out = softmax(concat(rs0,rs1,rs2) @ w_classify)
    classify_wmma_kernel<<<(M_TILES + 7) / 8, 256, 0, stream>>>(rs0, rs1, rs2, wcswz, outp);
}
